// TGN_53223234732237
// MI455X (gfx1250) — matrix operand order repaired and checked by reading, not compiled, not run on hardware
//
#include <hip/hip_runtime.h>

typedef __attribute__((ext_vector_type(16))) _Float16 v16h;
typedef __attribute__((ext_vector_type(8)))  _Float16 v8h;
typedef __attribute__((ext_vector_type(8)))  float    v8f;

#define MEM_D  500
#define MSG_D  100
#define EDGE_F 101
#define RAW_D  701
#define HID_D  350

// padded tile geometry (K chunks of 32, N tiles of 16)
#define KC1 22   // MLP1 K: 701 -> 704
#define CT1 22   // MLP1 N: 350 -> 352
#define KC2 11   // MLP2 K: 350 -> 352
#define CT2 8    // MLP2 N: 100 -> 128
#define KCI 4    // GRU gi K: 100 -> 128
#define KCH 16   // GRU gh K: 500 -> 512
#define CTG 96   // GRU packed N tiles: 3 gates * 512 / 16

// LDS row strides (halves): multiple of 8 (16B align) and bank-conflict free
#define RS 712
#define MS 360
#define GS 136

// ---------------------------------------------------------------------------
// device helpers
// ---------------------------------------------------------------------------
__device__ __forceinline__ v8f wmma_f16(v16h a, v16h b, v8f c) {
    // D = A(16x32 f16) * B(32x16 f16) + C(16x16 f32)
    return __builtin_amdgcn_wmma_f32_16x16x32_f16(false, a, false, b, (short)0, c,
                                                  false, false);
}

// A-fragment from LDS row-major tile (16 rows x K): lane<16 holds K 0..7,16..23,
// lane>=16 holds K 8..15,24..31 of its row (CDNA5 16-bit A layout).
__device__ __forceinline__ v16h afrag(const _Float16* tile, int stride, int kc, int lane) {
    const _Float16* p = tile + (lane & 15) * stride + kc * 32 + ((lane < 16) ? 0 : 8);
    v8h lo = *(const v8h*)p;
    v8h hi = *(const v8h*)(p + 16);
    v16h r;
#pragma unroll
    for (int i = 0; i < 8; ++i) { r[i] = lo[i]; r[i + 8] = hi[i]; }
    return r;
}

// B-fragment: pre-swizzled by the pack kernel; 32 contiguous bytes per lane.
__device__ __forceinline__ v16h bfrag(const _Float16* Wp, int chunk, int lane) {
    return *(const v16h*)(Wp + (size_t)chunk * 512 + (size_t)lane * 16);
}

// ---------------------------------------------------------------------------
// weight pre-pack: f32 [Kdim x Ndim] -> f16 WMMA B-fragment layout, zero padded.
// B layout (matches A fragment): lanes 0-15 hold K 0..7,16..23, lanes
// 16-31 hold K 8..15,24..31, 2 consecutive K per VGPR; column n = lane%16.
// gru==1: logical col n' = g*512 + c maps to source col g*500 + c (3 gates).
// ---------------------------------------------------------------------------
__global__ void tgn_pack(const float* __restrict__ W, _Float16* __restrict__ Wp,
                         int Kdim, int Ndim, int KC, int CT, int gru) {
    int idx = blockIdx.x * 256 + threadIdx.x;
    int total = CT * KC * 512;
    if (idx >= total) return;
    int j = idx & 15;
    int l = (idx >> 4) & 31;
    int chunk = idx >> 9;
    int kc = chunk % KC;
    int ct = chunk / KC;
    int n = ct * 16 + (l & 15);
    int k = kc * 32 + ((l < 16) ? 0 : 8) + ((j < 8) ? j : 8 + j);
    float v = 0.f;
    if (k < Kdim) {
        if (gru) {
            int g = n >> 9, c = n & 511;
            if (c < MEM_D) v = W[(long)k * 1500 + g * MEM_D + c];
        } else if (n < Ndim) {
            v = W[(long)k * Ndim + n];
        }
    }
    Wp[(long)chunk * 512 + l * 16 + j] = (_Float16)v;
}

// ---------------------------------------------------------------------------
// winner selection: key = (time << 32) | event_index, atomicMax per node.
// ---------------------------------------------------------------------------
__global__ void tgn_key_init(const int* __restrict__ nids,
                             unsigned long long* __restrict__ keys, int B) {
    int i = blockIdx.x * 256 + threadIdx.x;
    if (i < B) keys[nids[i]] = 0ull;
}

__global__ void tgn_key_max(const int* __restrict__ nids, const int* __restrict__ times,
                            unsigned long long* __restrict__ keys, int B) {
    int i = blockIdx.x * 256 + threadIdx.x;
    if (i < B) {
        unsigned long long k =
            ((unsigned long long)(unsigned)times[i] << 32) | (unsigned)i;
        atomicMax(&keys[nids[i]], k);
    }
}

// ---------------------------------------------------------------------------
// main fused kernel: 64 events per workgroup, 256 threads (8 wave32)
// ---------------------------------------------------------------------------
__global__ __launch_bounds__(256) void tgn_main(
    const float* __restrict__ memory, const float* __restrict__ last_update,
    const float* __restrict__ edge_feats, const float* __restrict__ w_t,
    const float* __restrict__ b_t, const float* __restrict__ b1,
    const float* __restrict__ b2, const float* __restrict__ b_ih,
    const float* __restrict__ b_hh, const float* __restrict__ defmem,
    const int* __restrict__ node_ids, const int* __restrict__ edge_times,
    const unsigned long long* __restrict__ keys, const _Float16* __restrict__ W1p,
    const _Float16* __restrict__ W2p, const _Float16* __restrict__ Wihp,
    const _Float16* __restrict__ Whhp, float* __restrict__ out, int B) {
    __shared__ __align__(16) _Float16 rawA[64 * RS];  // [h | ef | tenc | 0pad] f16
    __shared__ __align__(16) _Float16 midA[64 * MS];  // relu(raw@W1+b1) f16
    __shared__ __align__(16) _Float16 msgA[64 * GS];  // msg (padded to 128) f16
    __shared__ int   nidL[64];
    __shared__ float dtL[64];
    __shared__ int   winL[64];
    __shared__ int   defL[64];

    const int tid = threadIdx.x;
    const int wg = blockIdx.x;
    const int lane = tid & 31;
    const int wid = tid >> 5;

    // ---- phase A0: per-event scalars -------------------------------------
    if (tid < 64) {
        const int e = wg * 64 + tid;
        int nid = 0, win = 0, deff = 0;
        float dt = 0.f;
        if (e < B) {
            nid = node_ids[e];
            const float t = (float)edge_times[e];
            const float lu = last_update[nid];
            dt = t - lu;
            const unsigned long long key = keys[nid];
            win = ((unsigned)key == (unsigned)e);
            const float tmax = (float)(unsigned)(key >> 32);
            deff = (fmaxf(lu, tmax) == 0.0f);
        }
        nidL[tid] = nid; dtL[tid] = dt; winL[tid] = win; defL[tid] = deff;
    }
    __syncthreads();

    // ---- phase A1: build raw = [h(500) | ef(101) | tenc(100) | 0] in f16 --
    {
        const int mrow = tid >> 2, sub = tid & 3;
        const int e2 = wg * 64 + mrow;
        _Float16* rrow = &rawA[mrow * RS];
        if (e2 < B) {
            const long nid = nidL[mrow];
            const float4* m4 = (const float4*)(memory + nid * (long)MEM_D);
            for (int k4 = sub; k4 < 125; k4 += 4) {
                float4 v = m4[k4];
                rrow[k4 * 4 + 0] = (_Float16)v.x;
                rrow[k4 * 4 + 1] = (_Float16)v.y;
                rrow[k4 * 4 + 2] = (_Float16)v.z;
                rrow[k4 * 4 + 3] = (_Float16)v.w;
            }
            for (int k = sub; k < EDGE_F; k += 4)
                rrow[MEM_D + k] = (_Float16)edge_feats[(long)e2 * EDGE_F + k];
            const float dtv = dtL[mrow];
            for (int j = sub; j < 100; j += 4)
                rrow[601 + j] = (_Float16)cosf(fmaf(dtv, w_t[j], b_t[j]));
            if (sub == 0)
                for (int k = RAW_D; k < RS; ++k) rrow[k] = (_Float16)0.f;
        } else {
            for (int k = sub; k < RS; k += 4) rrow[k] = (_Float16)0.f;
        }
    }
    __syncthreads();

    const v8f z8 = {0.f, 0.f, 0.f, 0.f, 0.f, 0.f, 0.f, 0.f};
    const int rh = (lane >> 4) * 8;  // C-layout row offset: lanes 16-31 -> rows+8

    // ---- MLP1: mid = relu(raw @ W1 + b1), N tiles = 22 --------------------
    for (int ct = wid; ct < CT1; ct += 8) {
        v8f acc[4] = {z8, z8, z8, z8};
        for (int kc = 0; kc < KC1; ++kc) {
            v16h b = bfrag(W1p, ct * KC1 + kc, lane);
#pragma unroll
            for (int rt = 0; rt < 4; ++rt) {
                v16h a = afrag(&rawA[rt * 16 * RS], RS, kc, lane);
                acc[rt] = wmma_f16(a, b, acc[rt]);
            }
        }
        const int n = ct * 16 + (lane & 15);
        const float bias = (n < HID_D) ? b1[n] : 0.f;
        for (int rt = 0; rt < 4; ++rt)
#pragma unroll
            for (int v = 0; v < 8; ++v)
                midA[(rt * 16 + rh + v) * MS + n] =
                    (_Float16)fmaxf(acc[rt][v] + bias, 0.f);
    }
    __syncthreads();

    // ---- MLP2: msg = mid @ W2 + b2, N tiles = 8 (padded cols -> exact 0) --
    {
        const int ct = wid;  // 0..7
        v8f acc[4] = {z8, z8, z8, z8};
        for (int kc = 0; kc < KC2; ++kc) {
            v16h b = bfrag(W2p, ct * KC2 + kc, lane);
#pragma unroll
            for (int rt = 0; rt < 4; ++rt) {
                v16h a = afrag(&midA[rt * 16 * MS], MS, kc, lane);
                acc[rt] = wmma_f16(a, b, acc[rt]);
            }
        }
        const int n = ct * 16 + (lane & 15);
        const float bias = (n < MSG_D) ? b2[n] : 0.f;
        for (int rt = 0; rt < 4; ++rt)
#pragma unroll
            for (int v = 0; v < 8; ++v)
                msgA[(rt * 16 + rh + v) * GS + n] = (_Float16)(acc[rt][v] + bias);
    }
    __syncthreads();

    // ---- GRU: per column tile, all three gates in registers ---------------
    for (int ct = wid; ct < 32; ct += 8) {
        v8f aR[4] = {z8, z8, z8, z8}, aZ[4] = {z8, z8, z8, z8};
        v8f aNi[4] = {z8, z8, z8, z8}, aNh[4] = {z8, z8, z8, z8};
        // gi = msg @ W_ih (K = 128)
        for (int kc = 0; kc < KCI; ++kc) {
            v16h bR = bfrag(Wihp, (0 * 32 + ct) * KCI + kc, lane);
            v16h bZ = bfrag(Wihp, (1 * 32 + ct) * KCI + kc, lane);
            v16h bN = bfrag(Wihp, (2 * 32 + ct) * KCI + kc, lane);
#pragma unroll
            for (int rt = 0; rt < 4; ++rt) {
                v16h a = afrag(&msgA[rt * 16 * GS], GS, kc, lane);
                aR[rt] = wmma_f16(a, bR, aR[rt]);
                aZ[rt] = wmma_f16(a, bZ, aZ[rt]);
                aNi[rt] = wmma_f16(a, bN, aNi[rt]);
            }
        }
        // gh = h @ W_hh (K = 512; A cols 500..511 are junk but finite,
        // W_hh rows 500..511 are packed as exact zeros)
        for (int kc = 0; kc < KCH; ++kc) {
            v16h bR = bfrag(Whhp, (0 * 32 + ct) * KCH + kc, lane);
            v16h bZ = bfrag(Whhp, (1 * 32 + ct) * KCH + kc, lane);
            v16h bN = bfrag(Whhp, (2 * 32 + ct) * KCH + kc, lane);
#pragma unroll
            for (int rt = 0; rt < 4; ++rt) {
                v16h a = afrag(&rawA[rt * 16 * RS], RS, kc, lane);
                aR[rt] = wmma_f16(a, bR, aR[rt]);
                aZ[rt] = wmma_f16(a, bZ, aZ[rt]);
                aNh[rt] = wmma_f16(a, bN, aNh[rt]);
            }
        }
        // epilogue: gates in f32, exact f32 h, write winner rows only
        const int c = ct * 16 + (lane & 15);
        const bool cv = (c < MEM_D);
        float bR = 0.f, bZ = 0.f, bNi = 0.f, bNh = 0.f;
        if (cv) {
            bR = b_ih[c] + b_hh[c];
            bZ = b_ih[MEM_D + c] + b_hh[MEM_D + c];
            bNi = b_ih[2 * MEM_D + c];
            bNh = b_hh[2 * MEM_D + c];
        }
        for (int rt = 0; rt < 4; ++rt) {
#pragma unroll
            for (int v = 0; v < 8; ++v) {
                const int m = rt * 16 + rh + v;
                const int e = wg * 64 + m;
                if (cv && winL[m] && e < B) {
                    const float r = 1.f / (1.f + expf(-(aR[rt][v] + bR)));
                    const float z = 1.f / (1.f + expf(-(aZ[rt][v] + bZ)));
                    const float nn = tanhf(aNi[rt][v] + bNi + r * (aNh[rt][v] + bNh));
                    const float hv = memory[(long)nidL[m] * MEM_D + c];
                    float hn = (1.f - z) * nn + z * hv;
                    if (defL[m]) hn = defmem[c];
                    out[(long)e * MEM_D + c] = hn;
                }
            }
        }
    }
}

// ---------------------------------------------------------------------------
// fixup: non-winner rows copy from their node's winner row (already in out)
// ---------------------------------------------------------------------------
__global__ void tgn_fixup(const int* __restrict__ nids,
                          const unsigned long long* __restrict__ keys,
                          const float* __restrict__ last_update,
                          const float* __restrict__ defmem, float* __restrict__ out,
                          int B) {
    const int i = blockIdx.x;
    if (i >= B) return;
    const int nid = nids[i];
    const unsigned long long key = keys[nid];
    const unsigned w = (unsigned)key;
    if (w == (unsigned)i) return;  // winner row already written by tgn_main
    const float lu2 = fmaxf(last_update[nid], (float)(unsigned)(key >> 32));
    const bool def = (lu2 == 0.f);
    const float* src = out + (long)w * MEM_D;
    float* dst = out + (long)i * MEM_D;
    for (int c = threadIdx.x; c < MEM_D; c += 256) dst[c] = def ? defmem[c] : src[c];
}

// ---------------------------------------------------------------------------
// launcher
// ---------------------------------------------------------------------------
extern "C" void kernel_launch(void* const* d_in, const int* in_sizes, int n_in,
                              void* d_out, int out_size, void* d_ws, size_t ws_size,
                              hipStream_t stream) {
    const float* memory      = (const float*)d_in[0];
    const float* last_update = (const float*)d_in[1];
    const float* edge_feats  = (const float*)d_in[2];
    const float* w_t         = (const float*)d_in[3];
    const float* b_t         = (const float*)d_in[4];
    const float* W1          = (const float*)d_in[5];
    const float* b1          = (const float*)d_in[6];
    const float* W2          = (const float*)d_in[7];
    const float* b2          = (const float*)d_in[8];
    const float* W_ih        = (const float*)d_in[9];
    const float* W_hh        = (const float*)d_in[10];
    const float* b_ih        = (const float*)d_in[11];
    const float* b_hh        = (const float*)d_in[12];
    const float* defmem      = (const float*)d_in[13];
    const int*   node_ids    = (const int*)d_in[14];
    const int*   edge_times  = (const int*)d_in[15];
    float* out = (float*)d_out;
    const int B = in_sizes[14];
    const int n_nodes = in_sizes[1];

    char* ws = (char*)d_ws;
    size_t off = 0;
    auto a256 = [](size_t x) { return (x + 255) & ~(size_t)255; };
    unsigned long long* keys = (unsigned long long*)(ws + off);
    off = a256(off + (size_t)n_nodes * 8);
    _Float16* W1p = (_Float16*)(ws + off);
    off = a256(off + (size_t)CT1 * KC1 * 512 * 2);
    _Float16* W2p = (_Float16*)(ws + off);
    off = a256(off + (size_t)CT2 * KC2 * 512 * 2);
    _Float16* Wihp = (_Float16*)(ws + off);
    off = a256(off + (size_t)CTG * KCI * 512 * 2);
    _Float16* Whhp = (_Float16*)(ws + off);
    off = a256(off + (size_t)CTG * KCH * 512 * 2);

    // weight pre-pack (tiny; amortized across the whole GEMM pipeline)
    {
        int t1 = CT1 * KC1 * 512;
        tgn_pack<<<(t1 + 255) / 256, 256, 0, stream>>>(W1, W1p, RAW_D, HID_D, KC1, CT1, 0);
        int t2 = CT2 * KC2 * 512;
        tgn_pack<<<(t2 + 255) / 256, 256, 0, stream>>>(W2, W2p, HID_D, MSG_D, KC2, CT2, 0);
        int t3 = CTG * KCI * 512;
        tgn_pack<<<(t3 + 255) / 256, 256, 0, stream>>>(W_ih, Wihp, MSG_D, 1500, KCI, CTG, 1);
        int t4 = CTG * KCH * 512;
        tgn_pack<<<(t4 + 255) / 256, 256, 0, stream>>>(W_hh, Whhp, MEM_D, 1500, KCH, CTG, 1);
    }

    // per-node winner keys
    tgn_key_init<<<(B + 255) / 256, 256, 0, stream>>>(node_ids, keys, B);
    tgn_key_max<<<(B + 255) / 256, 256, 0, stream>>>(node_ids, edge_times, keys, B);

    // fused gather + MLP + GRU, winner rows -> out
    tgn_main<<<(B + 63) / 64, 256, 0, stream>>>(
        memory, last_update, edge_feats, w_t, b_t, b1, b2, b_ih, b_hh, defmem,
        node_ids, edge_times, keys, W1p, W2p, Wihp, Whhp, out, B);

    // broadcast winner rows to remaining events
    tgn_fixup<<<B, 256, 0, stream>>>(node_ids, keys, last_update, defmem, out, B);
}
